// Attention_64269890617453
// MI455X (gfx1250) — matrix operand order repaired and checked by reading, not compiled, not run on hardware
//
#include <hip/hip_runtime.h>
#include <hip/hip_bf16.h>

typedef __bf16 bf16_t;
typedef __attribute__((ext_vector_type(16))) __bf16 v16bf;
typedef __attribute__((ext_vector_type(8)))  __bf16 bf16x8;
typedef __attribute__((ext_vector_type(8)))  float  v8f;

// async-to-LDS builtin operand types: b128 takes GCC-vector int4 pointers,
// global source in AS1 ("__device__"), LDS dest in AS3 ("__shared__").
typedef int v4i_t __attribute__((vector_size(16)));
typedef __attribute__((address_space(1))) v4i_t gv4i_t;
typedef __attribute__((address_space(3))) v4i_t lv4i_t;

#define WMMA_BF16(a, b, c) \
  __builtin_amdgcn_wmma_f32_16x16x32_bf16(false, (a), false, (b), (short)0, (c), false, false)

// async direct-to-LDS 16B copy, tracked by ASYNCcnt
#define ASYNC_LD_B128(gsrc, ldst) \
  __builtin_amdgcn_global_load_async_to_lds_b128((gv4i_t*)(gsrc), (lv4i_t*)(ldst), 0, 0)

constexpr int   NROW  = 8192;
constexpr int   INF_  = 256;   // input feature dim
constexpr int   DF    = 128;   // output feature dim
constexpr float ALPHA = 0.2f;

// ---------------------------------------------------------------------------
// Kernel 1: q/k/v = relu(x @ W + b), f32 in -> bf16 out, via bf16 WMMA.
// Grid: (NROW/16, 2, 3); block 128 threads = 4 waves; wave -> 16x16 tile.
// ---------------------------------------------------------------------------
__global__ __launch_bounds__(128) void proj_relu_wmma(
    const float* __restrict__ x,
    const float* __restrict__ Wq, const float* __restrict__ bq,
    const float* __restrict__ Wk, const float* __restrict__ bk,
    const float* __restrict__ Wv, const float* __restrict__ bv,
    bf16_t* __restrict__ qout, bf16_t* __restrict__ kout, bf16_t* __restrict__ vout)
{
  const int lane = threadIdx.x & 31;
  const int wv   = threadIdx.x >> 5;
  const int half = lane >> 4;
  const int l16  = lane & 15;
  const int r0   = blockIdx.x * 16;
  const int d0   = (blockIdx.y * 4 + wv) * 16;
  const int z    = blockIdx.z;

  const float* W = (z == 0) ? Wq : (z == 1) ? Wk : Wv;
  const float* B = (z == 0) ? bq : (z == 1) ? bk : bv;
  bf16_t* dst    = (z == 0) ? qout : (z == 1) ? kout : vout;

  v8f acc = {};
  for (int kc = 0; kc < INF_ / 32; ++kc) {
    const float* xr = x + (size_t)(r0 + l16) * INF_ + kc * 32 + half * 16;
    v16bf a;
#pragma unroll
    for (int i = 0; i < 8; ++i) { a[i] = (bf16_t)xr[i]; a[8 + i] = (bf16_t)xr[8 + i]; }
    const float* wp = W + (size_t)(kc * 32 + half * 16) * DF + d0 + l16;
    v16bf b;
#pragma unroll
    for (int i = 0; i < 16; ++i) b[i] = (bf16_t)wp[(size_t)i * DF];
    acc = WMMA_BF16(a, b, acc);
  }
  const float bias = B[d0 + l16];
#pragma unroll
  for (int r = 0; r < 8; ++r) {
    float v = fmaxf(acc[r] + bias, 0.0f);
    dst[(size_t)(r0 + r + 8 * half) * DF + d0 + l16] = (bf16_t)v;
  }
}

// ---------------------------------------------------------------------------
// Kernel 2: fused leaky-relu + softmax attention, flash style, double-buffered
// async-to-LDS staging. 128 blocks x 4 waves; wave owns 16 query rows.
// ---------------------------------------------------------------------------
__global__ __launch_bounds__(128) void flash_attn_wmma(
    const bf16_t* __restrict__ qq, const bf16_t* __restrict__ kq,
    const bf16_t* __restrict__ vq, float* __restrict__ out)
{
  __shared__ bf16_t Kt[2][32 * 128];  // K tiles, row-major [key][feat]  (16 KB)
  __shared__ bf16_t Vt[2][128 * 32];  // V tiles, transposed [feat][key] (16 KB)
  __shared__ float  Pb[4][16 * 32];   // per-wave P relayout scratch      (8 KB)

  const int t    = threadIdx.x;
  const int lane = t & 31;
  const int wv   = t >> 5;
  const int half = lane >> 4;
  const int l16  = lane & 15;
  const int r0   = blockIdx.x * 64 + wv * 16;
  const int key  = t >> 2;   // 0..31
  const int seg  = t & 3;    // 32-feature segment

  // Q rows as 4 resident A-fragments (16x32 each)
  v16bf qa[4];
  {
    const bf16_t* qr = qq + (size_t)(r0 + l16) * DF;
#pragma unroll
    for (int dc = 0; dc < 4; ++dc) {
      const bf16_t* p = qr + dc * 32 + half * 16;
      v16bf a;
#pragma unroll
      for (int i = 0; i < 8; ++i) { a[i] = p[i]; a[8 + i] = p[8 + i]; }
      qa[dc] = a;
    }
  }

  v8f O[8];
#pragma unroll
  for (int i = 0; i < 8; ++i) O[i] = {};
  float mrow[8], lrow[8];
#pragma unroll
  for (int r = 0; r < 8; ++r) { mrow[r] = -3.0e38f; lrow[r] = 0.0f; }

  // ---- prologue: stage tile 0 into buffer 0 ----
  {
    const bf16_t* ks = kq + (size_t)key * DF + seg * 32;
    bf16_t*       kd = &Kt[0][key * 128 + seg * 32];
#pragma unroll
    for (int c = 0; c < 4; ++c) ASYNC_LD_B128(ks + c * 8, kd + c * 8);
    const bf16x8* vs = (const bf16x8*)(vq + (size_t)key * DF + seg * 32);
    bf16x8 vr[4];
#pragma unroll
    for (int c = 0; c < 4; ++c) vr[c] = vs[c];
#pragma unroll
    for (int c = 0; c < 4; ++c)
#pragma unroll
      for (int e = 0; e < 8; ++e)
        Vt[0][(seg * 32 + c * 8 + e) * 32 + key] = vr[c][e];
    asm volatile("s_wait_asynccnt 0" ::: "memory");
  }
  __syncthreads();

  for (int j0 = 0; j0 < NROW; j0 += 32) {
    const int  cur      = (j0 >> 5) & 1;
    const int  nxt      = cur ^ 1;
    const bool has_next = (j0 + 32) < NROW;

    // ---- kick off staging for the NEXT tile (overlaps with compute) ----
    bf16x8 vr[4];
    if (has_next) {
      const bf16_t* ks = kq + (size_t)(j0 + 32 + key) * DF + seg * 32;
      bf16_t*       kd = &Kt[nxt][key * 128 + seg * 32];
#pragma unroll
      for (int c = 0; c < 4; ++c) ASYNC_LD_B128(ks + c * 8, kd + c * 8);
      const bf16x8* vs = (const bf16x8*)(vq + (size_t)(j0 + 32 + key) * DF + seg * 32);
#pragma unroll
      for (int c = 0; c < 4; ++c) vr[c] = vs[c];  // in flight during compute
    }

    // ---- S = Q K^T for 32 keys: two 16x16 C tiles, K-dim 128 = 4 x 32 ----
    v8f S0 = {}, S1 = {};
#pragma unroll
    for (int dc = 0; dc < 4; ++dc) {
      const bf16_t* p0 = &Kt[cur][l16 * 128 + dc * 32 + half * 16];
      const bf16_t* p1 = &Kt[cur][(l16 + 16) * 128 + dc * 32 + half * 16];
      v16bf b0, b1;
#pragma unroll
      for (int i = 0; i < 16; ++i) { b0[i] = p0[i]; b1[i] = p1[i]; }
      S0 = WMMA_BF16(qa[dc], b0, S0);
      S1 = WMMA_BF16(qa[dc], b1, S1);
    }

    // ---- LeakyReLU + online softmax (row r lives in (VGPR r, lane-half)) ----
    float corr[8];
#pragma unroll
    for (int r = 0; r < 8; ++r) {
      float s0 = S0[r]; s0 = (s0 > 0.f) ? s0 : ALPHA * s0;
      float s1 = S1[r]; s1 = (s1 > 0.f) ? s1 : ALPHA * s1;
      float mx = fmaxf(s0, s1);
      mx = fmaxf(mx, __shfl_xor(mx, 1, 32));
      mx = fmaxf(mx, __shfl_xor(mx, 2, 32));
      mx = fmaxf(mx, __shfl_xor(mx, 4, 32));
      mx = fmaxf(mx, __shfl_xor(mx, 8, 32));
      const float mnew = fmaxf(mrow[r], mx);
      const float c    = __expf(mrow[r] - mnew);
      mrow[r] = mnew;
      const float p0 = __expf(s0 - mnew);
      const float p1 = __expf(s1 - mnew);
      float rs = p0 + p1;
      rs += __shfl_xor(rs, 1, 32);
      rs += __shfl_xor(rs, 2, 32);
      rs += __shfl_xor(rs, 4, 32);
      rs += __shfl_xor(rs, 8, 32);
      lrow[r] = lrow[r] * c + rs;
      corr[r] = c;
      S0[r] = p0; S1[r] = p1;
    }
#pragma unroll
    for (int tt = 0; tt < 8; ++tt)
#pragma unroll
      for (int r = 0; r < 8; ++r) O[tt][r] *= corr[r];

    // ---- P (C layout) -> per-wave LDS -> bf16 A-fragment ----
    float* pb = &Pb[wv][0];
#pragma unroll
    for (int r = 0; r < 8; ++r) {
      pb[(r + 8 * half) * 32 + l16]      = S0[r];
      pb[(r + 8 * half) * 32 + 16 + l16] = S1[r];
    }
    asm volatile("s_wait_dscnt 0" ::: "memory");  // same-wave LDS is in order
    v16bf pa;
    {
      const float* pr = pb + l16 * 32 + half * 16;
#pragma unroll
      for (int i = 0; i < 8; ++i) { pa[i] = (bf16_t)pr[i]; pa[8 + i] = (bf16_t)pr[8 + i]; }
    }

    // ---- O += P V : 8 n-tiles across D=128 ----
#pragma unroll
    for (int tt = 0; tt < 8; ++tt) {
      const bf16_t* vp = &Vt[cur][(tt * 16 + l16) * 32 + half * 16];
      v16bf b;
#pragma unroll
      for (int i = 0; i < 16; ++i) b[i] = vp[i];
      O[tt] = WMMA_BF16(pa, b, O[tt]);
    }

    // ---- finish staging next tile: transposed V store, wait async K ----
    if (has_next) {
#pragma unroll
      for (int c = 0; c < 4; ++c)
#pragma unroll
        for (int e = 0; e < 8; ++e)
          Vt[nxt][(seg * 32 + c * 8 + e) * 32 + key] = vr[c][e];
    }
    asm volatile("s_wait_asynccnt 0" ::: "memory");
    __syncthreads();
  }

  // ---- epilogue: divide by row sums, write f32 output ----
#pragma unroll
  for (int r = 0; r < 8; ++r) {
    const float  inv = 1.0f / lrow[r];
    const size_t row = (size_t)(r0 + r + 8 * half);
#pragma unroll
    for (int tt = 0; tt < 8; ++tt)
      out[row * DF + tt * 16 + l16] = O[tt][r] * inv;
  }
}

// ---------------------------------------------------------------------------
extern "C" void kernel_launch(void* const* d_in, const int* in_sizes, int n_in,
                              void* d_out, int out_size, void* d_ws, size_t ws_size,
                              hipStream_t stream) {
  (void)in_sizes; (void)n_in; (void)out_size; (void)ws_size;
  const float* x  = (const float*)d_in[0];
  const float* Wq = (const float*)d_in[1];
  const float* bq = (const float*)d_in[2];
  const float* Wk = (const float*)d_in[3];
  const float* bk = (const float*)d_in[4];
  const float* Wv = (const float*)d_in[5];
  const float* bv = (const float*)d_in[6];
  float* out = (float*)d_out;

  bf16_t* qb = (bf16_t*)d_ws;                    // 2 MB
  bf16_t* kb = qb + (size_t)NROW * DF;           // 2 MB
  bf16_t* vb = kb + (size_t)NROW * DF;           // 2 MB  (ws >= 6 MB)

  proj_relu_wmma<<<dim3(NROW / 16, 2, 3), 128, 0, stream>>>(
      x, Wq, bq, Wk, bk, Wv, bv, qb, kb, vb);
  flash_attn_wmma<<<dim3(NROW / 64), 128, 0, stream>>>(qb, kb, vb, out);
}
